// MultiHeadAttentionQuantum_65481071407275
// MI455X (gfx1250) — matrix operand order repaired and checked by reading, not compiled, not run on hardware
//
#include <hip/hip_runtime.h>
#include <hip/hip_bf16.h>
#include <math.h>

// ---------------------------------------------------------------------------
// MultiHeadAttentionQuantum for MI455X (gfx1250, wave32, WMMA f16 16x16x32)
//   B=2, S=2048, E=1024, H=16, DK=64
//   out = softmax( cos(x Wq^T + theta) (x Wk^T)^T / 8 ) (x Wv^T)
// Pipeline:
//   1) f32 -> f16 packing of x, Wk, Wv, Wq
//   2) WMMA GEMM (x_h @ W_h^T) with per-mode epilogue:
//        mode 0: K  -> kbuf  [bh, s, d]   (f16)
//        mode 1: V  -> vtbuf [bh, d, s]   (f16, transposed for PV gemm)
//        mode 2: Q  -> qbuf  [bh, s, d] = f16(cos(acc + theta[d]))
//   3) flash attention: 8 waves/WG share double-buffered K/V^T blocks staged
//      into LDS with GLOBAL_LOAD_ASYNC_TO_LDS_B128 (ASYNCcnt), all matrix math
//      on v_wmma_f32_16x16x32_f16 in transposed-score formulation.
// ---------------------------------------------------------------------------

typedef __attribute__((ext_vector_type(16))) _Float16 v16h;
typedef __attribute__((ext_vector_type(8)))  _Float16 v8h;
typedef __attribute__((ext_vector_type(4)))  _Float16 v4h;
typedef __attribute__((ext_vector_type(8)))  float    v8f;

#define BQ   2
#define SQ   2048
#define EQ   1024
#define HQ   16
#define DKQ  64
#define MROWS (BQ*SQ)          // 4096

// padded LDS row strides (halves) -> distinct banks for the 16 row-readers
#define KPAD 72                // 32 rows x 72 halves (K block, row = key t)
#define VPAD 40                // 64 rows x 40 halves (V^T block, row = d)

static __device__ __forceinline__ v8f wmma16x16x32(v16h a, v16h b, v8f c) {
  // (neg_a, A, neg_b, B, c_mod, C, reuse_a, reuse_b)
  return __builtin_amdgcn_wmma_f32_16x16x32_f16(false, a, false, b, (short)0, c,
                                                false, false);
}

// Low 32 bits of a flat LDS-aperture address are the LDS byte offset
// (CDNA5 ISA 10.2: aperture selected by addr[63:32]).
static __device__ __forceinline__ unsigned ldsOff(const void* p) {
  return (unsigned)(uintptr_t)p;
}

// Async copy 16B global -> LDS, tracked by ASYNCcnt (CDNA5 ISA 10.7 / 15.18.3).
static __device__ __forceinline__ void async_g2l_b128(const void* gptr,
                                                      unsigned lds_byte_off) {
  asm volatile("global_load_async_to_lds_b128 %0, %1, off"
               :: "v"(lds_byte_off), "v"(gptr) : "memory");
}
static __device__ __forceinline__ void wait_asynccnt0() {
  asm volatile("s_wait_asynccnt 0x0" ::: "memory");
}

// -------------------------- f32 -> f16 pack --------------------------------
__global__ void cvt_f32_to_f16(const float* __restrict__ src,
                               _Float16* __restrict__ dst, int n4) {
  int i = blockIdx.x * blockDim.x + threadIdx.x;
  if (i < n4) {
    const float4 f = ((const float4*)src)[i];
    v4h o;
    o[0] = (_Float16)f.x; o[1] = (_Float16)f.y;
    o[2] = (_Float16)f.z; o[3] = (_Float16)f.w;
    ((v4h*)dst)[i] = o;
  }
}

// -------------------------- projection GEMM --------------------------------
// C[m][n] = sum_k X[m][k] * W[n][k];  M=4096, N=1024, K=1024
// block = 256 threads (8 waves). Block tile 128(M) x 64(N).
// wave tile 16(M) x 64(N) -> 4 wmma accumulators, K-loop step 32.
__global__ __launch_bounds__(256)
void gemm_proj_wmma(const _Float16* __restrict__ X,
                    const _Float16* __restrict__ W,
                    _Float16* __restrict__ outb,
                    const float* __restrict__ theta,
                    int mode) {
  const int lane  = threadIdx.x & 31;
  const int wave  = threadIdx.x >> 5;
  const int n     = lane & 15;       // column-in-tile (B/C) or row-in-tile (A)
  const int half  = lane >> 4;
  const int mtile = blockIdx.x * 128 + wave * 16;
  const int ntile = blockIdx.y * 64;

  v8f acc[4] = {v8f{}, v8f{}, v8f{}, v8f{}};

  const _Float16* xrow = X + (size_t)(mtile + n) * EQ;  // A: row m = mtile + n

  for (int kb = 0; kb < EQ; kb += 32) {
    // A fragment (16x32 f16): lane(half) -> K run [half*16..+15]
    v8h lo = *(const v8h*)(xrow + kb + half * 16);
    v8h hi = *(const v8h*)(xrow + kb + half * 16 + 8);
    v16h a;
#pragma unroll
    for (int i = 0; i < 8; ++i) { a[i] = lo[i]; a[i + 8] = hi[i]; }

#pragma unroll
    for (int nt = 0; nt < 4; ++nt) {
      // B fragment (32x16): lanes 0-15 K=0..15, lanes 16-31 K=16..31;
      // B[k][n] = W[ntile+nt*16+n][kb+k] -> contiguous 16 halves.
      const _Float16* wrow = W + (size_t)(ntile + nt * 16 + n) * EQ;
      v16h bfr = *(const v16h*)(wrow + kb + half * 16);
      acc[nt] = wmma16x16x32(a, bfr, acc[nt]);
    }
  }

  // Epilogue. C layout: reg r, lanes 0-15 -> M=r, lanes 16-31 -> M=r+8; N=lane&15.
#pragma unroll
  for (int nt = 0; nt < 4; ++nt) {
    const int nglob = ntile + nt * 16 + n;
    const int h  = nglob >> 6;     // /64
    const int d  = nglob & 63;
    float th = 0.f;
    if (mode == 2) th = theta[d];
#pragma unroll
    for (int r = 0; r < 8; ++r) {
      const int mglob = mtile + r + 8 * half;
      const int bb = mglob >> 11;          // /2048
      const int s  = mglob & 2047;
      float v = acc[nt][r];
      if (mode == 2) v = __cosf(v + th);
      size_t idx;
      if (mode == 1)  // V transposed: [bh, d, s]
        idx = ((size_t)(bb * HQ + h) * DKQ + d) * SQ + s;
      else            // K / Q: [bh, s, d]
        idx = ((size_t)(bb * HQ + h) * SQ + s) * DKQ + d;
      outb[idx] = (_Float16)v;
    }
  }
}

// -------------------------- flash attention --------------------------------
// 8 waves / workgroup, one (b,h) per workgroup slice, 16 query rows per wave.
// K/V^T blocks (32 keys) staged into LDS once per workgroup via async copies,
// double-buffered. Transposed-score formulation:
//   scores^T[t][s] = sum_d K[t][d] Q[s][d]   (A = K block, B = Q^T)
//   out^T[d][s]   += sum_t V^T[d][t] P^T[t][s]
// so softmax row-stats live per lane (one query column per lane).
__global__ __launch_bounds__(256)
void flash_attn_wmma(const _Float16* __restrict__ Q,
                     const _Float16* __restrict__ K,
                     const _Float16* __restrict__ VT,
                     float* __restrict__ out) {
  __shared__ _Float16 lsK[2][32 * KPAD];
  __shared__ _Float16 lsV[2][64 * VPAD];

  const int tid  = threadIdx.x;
  const int lane = tid & 31;
  const int wv   = tid >> 5;
  const int wg   = blockIdx.x * 8 + wv;       // 0 .. 4095
  const int qb   = wg & 127;                  // S/16 blocks
  const int bh   = wg >> 7;                   // 0 .. 31 (uniform per block)
  const int b    = bh >> 4;
  const int h    = bh & 15;
  const int sblk = qb * 16;
  const int n    = lane & 15;
  const int half = lane >> 4;

  const _Float16* qbase = Q  + ((size_t)bh * SQ + sblk) * DKQ;
  const _Float16* kbase = K  + (size_t)bh * SQ * DKQ;
  const _Float16* vbase = VT + (size_t)bh * DKQ * SQ;

  // per-thread staging assignment (256 threads, 16B chunks)
  const int kr = tid >> 3, kc = tid & 7;      // K: 32 rows x 8 chunks
  const int vr = tid >> 2, vc = tid & 3;      // V: 64 rows x 4 chunks

  // Q B-fragments (reused for all key blocks): b[j] = Q[sblk+n][dc*32+half*16+j]
  v16h qf[2];
#pragma unroll
  for (int dc = 0; dc < 2; ++dc)
    qf[dc] = *(const v16h*)(qbase + (size_t)n * DKQ + dc * 32 + half * 16);

  v8f acc[4] = {v8f{}, v8f{}, v8f{}, v8f{}};
  float m_run = -INFINITY, l_run = 0.f;

  // prologue: stage block 0 into buffer 0
  async_g2l_b128(kbase + (size_t)(0 + kr) * DKQ + kc * 8,
                 ldsOff(&lsK[0][kr * KPAD + kc * 8]));
  async_g2l_b128(vbase + (size_t)vr * SQ + 0 + vc * 8,
                 ldsOff(&lsV[0][vr * VPAD + vc * 8]));

  for (int i = 0; i < SQ / 32; ++i) {
    const int kb  = i * 32;
    const int buf = i & 1;

    wait_asynccnt0();      // my async fills done
    __syncthreads();       // everyone's fills done / prev reads of nxt done

    if (i + 1 < SQ / 32) { // prefetch next block into other buffer
      const int nb = (i + 1) & 1;
      async_g2l_b128(kbase + (size_t)(kb + 32 + kr) * DKQ + kc * 8,
                     ldsOff(&lsK[nb][kr * KPAD + kc * 8]));
      async_g2l_b128(vbase + (size_t)vr * SQ + (kb + 32) + vc * 8,
                     ldsOff(&lsV[nb][vr * VPAD + vc * 8]));
    }

    const _Float16* ksm = &lsK[buf][0];
    const _Float16* vsm = &lsV[buf][0];

    // ---- scores^T: two 16(t) x 16(s) tiles, K-dim d = 64 (2 steps) ----
    v8f s0 = v8f{}, s1 = v8f{};
#pragma unroll
    for (int dc = 0; dc < 2; ++dc) {
      const _Float16* kr0 = ksm + n * KPAD + dc * 32;
      v8h l0 = *(const v8h*)(kr0 + half * 16);
      v8h h0 = *(const v8h*)(kr0 + half * 16 + 8);
      v16h a0;
#pragma unroll
      for (int j = 0; j < 8; ++j) { a0[j] = l0[j]; a0[j + 8] = h0[j]; }
      s0 = wmma16x16x32(a0, qf[dc], s0);

      const _Float16* kr1 = ksm + (16 + n) * KPAD + dc * 32;
      v8h l1 = *(const v8h*)(kr1 + half * 16);
      v8h h1 = *(const v8h*)(kr1 + half * 16 + 8);
      v16h a1;
#pragma unroll
      for (int j = 0; j < 8; ++j) { a1[j] = l1[j]; a1[j + 8] = h1[j]; }
      s1 = wmma16x16x32(a1, qf[dc], s1);
    }

    // ---- online softmax over the 32 keys of this block ----
    float t0[8], t1[8];
    float mb = -INFINITY;
#pragma unroll
    for (int r = 0; r < 8; ++r) {
      t0[r] = s0[r] * 0.125f;          // 1/sqrt(64)
      t1[r] = s1[r] * 0.125f;
      mb = fmaxf(mb, fmaxf(t0[r], t1[r]));
    }
    mb = fmaxf(mb, __shfl_xor(mb, 16, 32));   // combine lane halves (same s)
    const float m_new = fmaxf(m_run, mb);
    const float scale = __expf(m_run - m_new);
    float rs = 0.f;
#pragma unroll
    for (int r = 0; r < 8; ++r) {
      t0[r] = __expf(t0[r] - m_new);
      t1[r] = __expf(t1[r] - m_new);
      rs += t0[r] + t1[r];
    }
    rs += __shfl_xor(rs, 16, 32);
    l_run = l_run * scale + rs;
    m_run = m_new;
#pragma unroll
    for (int dt = 0; dt < 4; ++dt)
#pragma unroll
      for (int r = 0; r < 8; ++r) acc[dt][r] *= scale;

    // ---- build P^T B-fragment from the C-layout tiles (lane-half swaps) ----
    v16h pb;
#pragma unroll
    for (int r = 0; r < 8; ++r) {
      const float c0 = __shfl_xor(t0[r], 16, 32);
      const float c1 = __shfl_xor(t1[r], 16, 32);
      pb[r]     = (_Float16)((half == 0) ? t0[r] : c1);
      pb[r + 8] = (_Float16)((half == 0) ? c0 : t1[r]);
    }

    // ---- out^T[d][s] += V^T (A) x P^T (B) ----
#pragma unroll
    for (int dt = 0; dt < 4; ++dt) {
      const _Float16* vrp = vsm + (dt * 16 + n) * VPAD;
      v8h lo = *(const v8h*)(vrp + half * 16);
      v8h hi = *(const v8h*)(vrp + half * 16 + 8);
      v16h va;
#pragma unroll
      for (int j = 0; j < 8; ++j) { va[j] = lo[j]; va[j + 8] = hi[j]; }
      acc[dt] = wmma16x16x32(va, pb, acc[dt]);
    }
  }

  // ---- finalize: out[b][s][h*64+d] = acc^T / l ----
  const float inv_l = 1.0f / l_run;
  const int s = sblk + n;
#pragma unroll
  for (int dt = 0; dt < 4; ++dt) {
#pragma unroll
    for (int r = 0; r < 8; ++r) {
      const int d = dt * 16 + r + 8 * half;
      out[(((size_t)b * SQ + s) * HQ + h) * DKQ + d] = acc[dt][r] * inv_l;
    }
  }
}

// ---------------------------------------------------------------------------
extern "C" void kernel_launch(void* const* d_in, const int* in_sizes, int n_in,
                              void* d_out, int out_size, void* d_ws, size_t ws_size,
                              hipStream_t stream) {
  const float* x     = (const float*)d_in[0];   // [B,S,E]
  const float* Wk    = (const float*)d_in[1];   // [E,E]
  const float* Wv    = (const float*)d_in[2];   // [E,E]
  const float* Wq    = (const float*)d_in[3];   // [H,DK,E] == [E,E] flat
  const float* theta = (const float*)d_in[4];   // [DK]

  char* ws = (char*)d_ws;
  const size_t XH  = (size_t)MROWS * EQ * 2;          // 8 MiB
  const size_t WH  = (size_t)EQ * EQ * 2;             // 2 MiB
  const size_t QKV = (size_t)BQ * HQ * SQ * DKQ * 2;  // 8 MiB
  _Float16* xh    = (_Float16*)(ws);
  _Float16* whk   = (_Float16*)(ws + XH);
  _Float16* whv   = (_Float16*)(ws + XH + WH);
  _Float16* whq   = (_Float16*)(ws + XH + 2 * WH);
  _Float16* kbuf  = (_Float16*)(ws + XH + 3 * WH);
  _Float16* vtbuf = (_Float16*)(ws + XH + 3 * WH + QKV);
  _Float16* qbuf  = (_Float16*)(ws + XH + 3 * WH + 2 * QKV);

  // 1) pack to f16
  {
    int n4 = MROWS * EQ / 4;
    cvt_f32_to_f16<<<(n4 + 255) / 256, 256, 0, stream>>>(x, xh, n4);
    int w4 = EQ * EQ / 4;
    cvt_f32_to_f16<<<(w4 + 255) / 256, 256, 0, stream>>>(Wk, whk, w4);
    cvt_f32_to_f16<<<(w4 + 255) / 256, 256, 0, stream>>>(Wv, whv, w4);
    cvt_f32_to_f16<<<(w4 + 255) / 256, 256, 0, stream>>>(Wq, whq, w4);
  }

  // 2) projections (WMMA GEMMs)
  dim3 ggrid(MROWS / 128, EQ / 64);
  gemm_proj_wmma<<<ggrid, 256, 0, stream>>>(xh, whk, kbuf,  theta, 0);
  gemm_proj_wmma<<<ggrid, 256, 0, stream>>>(xh, whv, vtbuf, theta, 1);
  gemm_proj_wmma<<<ggrid, 256, 0, stream>>>(xh, whq, qbuf,  theta, 2);

  // 3) flash attention (4096 waves, 8 waves/block sharing LDS-staged K/V)
  flash_attn_wmma<<<BQ * HQ * (SQ / 16) / 8, 256, 0, stream>>>(
      qbuf, kbuf, vtbuf, (float*)d_out);
}
